// HiDreamImageSingleTransformerBlock_74844100100787
// MI455X (gfx1250) — matrix operand order repaired and checked by reading, not compiled, not run on hardware
//
#include <hip/hip_runtime.h>
#include <hip/hip_bf16.h>

#define DIMX 2048
#define SEQ  2048
#define NHEAD 16
#define DHEAD 128
#define NEXP 4
#define HIDX 5632
#define HSX  2816
#define MOD6 (6 * DIMX)

typedef __attribute__((ext_vector_type(16))) __bf16 bf16x16;
typedef __attribute__((ext_vector_type(8)))  float  f32x8;

// round-to-nearest-even bf16, result in high 16 bits of the returned dword
static __device__ __forceinline__ unsigned bfr(float f) {
  unsigned u = __float_as_uint(f);
  return u + 0x7FFFu + ((u >> 16) & 1u);
}
static __device__ __forceinline__ unsigned short f2bf(float f) {
  return (unsigned short)(bfr(f) >> 16);
}
// pack two floats into {bf16(hi), bf16(lo)} with one v_perm_b32
static __device__ __forceinline__ unsigned pk2(float lo, float hi) {
  return __builtin_amdgcn_perm(bfr(hi), bfr(lo), 0x07060302u);
}
static __device__ __forceinline__ uint2 pack4(float4 v) {
  uint2 r;
  r.x = pk2(v.x, v.y);
  r.y = pk2(v.z, v.w);
  return r;
}

static __device__ __forceinline__ void ld_pair(const unsigned short* p, bf16x16& f, int idx) {
  union { unsigned u; __bf16 h[2]; } c;
  c.u = *(const unsigned*)(p);
  f[idx] = c.h[0];
  f[idx + 1] = c.h[1];
}

// K index base for 16-bit A-matrix fragment (16x32), contiguous k = 16*half + s.
static __device__ __forceinline__ int a_kbase(int r, int half) {
  return 2 * r + half * 16;
}

// ---------------------------------------------------------------------------
// bf16 WMMA GEMM: C[M,N] = A[M,K] * B[K,N] (+ bias), f32 in/out.
// Block tile 64x128, 8 waves (4x2); each wave: 16 rows x 64 cols = 4 WMMA C
// tiles per K-step. Double-buffered LDS: next-tile b128 global loads overlap
// the current tile's WMMAs; one barrier per K-step.
// ---------------------------------------------------------------------------
#define AST 40
#define BST 40
#define BLKN 128
__global__ __launch_bounds__(256)
void gemm_bf16(const float* __restrict__ A, int lda,
               const float* __restrict__ B, int ldb,
               const float* __restrict__ bias,
               float* __restrict__ C, int ldc, int Kdim) {
  __shared__ unsigned short As[2][64 * AST];     // [row][k]
  __shared__ unsigned short Bs[2][BLKN * BST];   // transposed [n][k]
  const int tid  = threadIdx.x;
  const int lane = tid & 31;
  const int wave = tid >> 5;
  const int half = lane >> 4;
  const int nl   = lane & 15;
  const int row0 = blockIdx.y * 64;
  const int col0 = blockIdx.x * BLKN;
  const int wm = (wave & 3) * 16;
  const int wn = (wave >> 2) * 64;

  // per-thread staging map: A: 8 consecutive k; B: 16 consecutive n
  const int arow = tid >> 2, akq = (tid & 3) * 8;
  const int bk   = tid >> 3, bnq = (tid & 7) * 16;
  const float* aP = A + (size_t)(row0 + arow) * lda + akq;
  const float* bP = B + (size_t)bk * ldb + col0 + bnq;
  const size_t bStep = (size_t)32 * ldb;
  const int asOff = arow * AST + akq;

  f32x8 acc[4];
  #pragma unroll
  for (int f = 0; f < 4; ++f)
    #pragma unroll
    for (int i = 0; i < 8; ++i) acc[f][i] = 0.f;

  // prologue: stage tile 0 into buffer 0
  {
    float4 a0 = *(const float4*)(aP);
    float4 a1 = *(const float4*)(aP + 4);
    *(uint2*)(&As[0][asOff])     = pack4(a0);
    *(uint2*)(&As[0][asOff + 4]) = pack4(a1);
    #pragma unroll
    for (int j = 0; j < 4; ++j) {
      float4 bv = *(const float4*)(bP + 4 * j);
      Bs[0][(bnq + 4 * j + 0) * BST + bk] = f2bf(bv.x);
      Bs[0][(bnq + 4 * j + 1) * BST + bk] = f2bf(bv.y);
      Bs[0][(bnq + 4 * j + 2) * BST + bk] = f2bf(bv.z);
      Bs[0][(bnq + 4 * j + 3) * BST + bk] = f2bf(bv.w);
    }
  }
  __syncthreads();

  int buf = 0;
  for (int kb = 0; kb < Kdim; kb += 32) {
    const bool hasNext = (kb + 32 < Kdim);
    float4 na0, na1, nb0, nb1, nb2, nb3;
    if (hasNext) {
      aP += 32;
      bP += bStep;
      na0 = *(const float4*)(aP);
      na1 = *(const float4*)(aP + 4);
      nb0 = *(const float4*)(bP);
      nb1 = *(const float4*)(bP + 4);
      nb2 = *(const float4*)(bP + 8);
      nb3 = *(const float4*)(bP + 12);
    }

    // compute from current buffer
    bf16x16 af;
    #pragma unroll
    for (int r = 0; r < 8; ++r)
      ld_pair(&As[buf][(wm + nl) * AST + a_kbase(r, half)], af, 2 * r);
    #pragma unroll
    for (int f = 0; f < 4; ++f) {
      bf16x16 bfv;
      #pragma unroll
      for (int r = 0; r < 8; ++r)
        ld_pair(&Bs[buf][(wn + f * 16 + nl) * BST + 2 * r + half * 16], bfv, 2 * r);
      acc[f] = __builtin_amdgcn_wmma_f32_16x16x32_bf16(false, af, false, bfv, (short)0, acc[f], false, false);
    }

    // stage next tile into the other buffer
    if (hasNext) {
      const int nb = buf ^ 1;
      *(uint2*)(&As[nb][asOff])     = pack4(na0);
      *(uint2*)(&As[nb][asOff + 4]) = pack4(na1);
      Bs[nb][(bnq + 0)  * BST + bk] = f2bf(nb0.x);
      Bs[nb][(bnq + 1)  * BST + bk] = f2bf(nb0.y);
      Bs[nb][(bnq + 2)  * BST + bk] = f2bf(nb0.z);
      Bs[nb][(bnq + 3)  * BST + bk] = f2bf(nb0.w);
      Bs[nb][(bnq + 4)  * BST + bk] = f2bf(nb1.x);
      Bs[nb][(bnq + 5)  * BST + bk] = f2bf(nb1.y);
      Bs[nb][(bnq + 6)  * BST + bk] = f2bf(nb1.z);
      Bs[nb][(bnq + 7)  * BST + bk] = f2bf(nb1.w);
      Bs[nb][(bnq + 8)  * BST + bk] = f2bf(nb2.x);
      Bs[nb][(bnq + 9)  * BST + bk] = f2bf(nb2.y);
      Bs[nb][(bnq + 10) * BST + bk] = f2bf(nb2.z);
      Bs[nb][(bnq + 11) * BST + bk] = f2bf(nb2.w);
      Bs[nb][(bnq + 12) * BST + bk] = f2bf(nb3.x);
      Bs[nb][(bnq + 13) * BST + bk] = f2bf(nb3.y);
      Bs[nb][(bnq + 14) * BST + bk] = f2bf(nb3.z);
      Bs[nb][(bnq + 15) * BST + bk] = f2bf(nb3.w);
    }
    __syncthreads();
    buf ^= 1;
  }

  #pragma unroll
  for (int f = 0; f < 4; ++f) {
    #pragma unroll
    for (int r = 0; r < 8; ++r) {
      int row = row0 + wm + r + half * 8;
      int col = col0 + wn + f * 16 + nl;
      float b = bias ? bias[col] : 0.f;
      C[(size_t)row * ldc + col] = acc[f][r] + b;
    }
  }
}

// ---------------------------------------------------------------------------
// Flash attention: one wave handles 16 query rows of one head.
// QK^T and AV via v_wmma_f32_16x16x32_bf16; online softmax in registers.
// ---------------------------------------------------------------------------
#define KST  136
#define VTST 40
#define PST  40
__global__ __launch_bounds__(32)
void attn_fa(const float* __restrict__ Q, const float* __restrict__ K,
             const float* __restrict__ V, float* __restrict__ O) {
  __shared__ unsigned short Ks[32 * KST];     // [key][dh] bf16
  __shared__ unsigned short Vt[128 * VTST];   // [dh][key] bf16 (transposed)
  __shared__ unsigned short Ps[16 * PST];     // [m][key] probs bf16
  const int lane = threadIdx.x;
  const int half = lane >> 4;
  const int nl   = lane & 15;
  const int h  = blockIdx.y;
  const int q0 = blockIdx.x * 16;
  const float scale = 0.08838834764831845f;   // 1/sqrt(128)

  bf16x16 qf[4];
  #pragma unroll
  for (int c = 0; c < 4; ++c) {
    #pragma unroll
    for (int r = 0; r < 8; ++r) {
      int dh = c * 32 + a_kbase(r, half);
      const float* qp = Q + (size_t)(q0 + nl) * DIMX + h * DHEAD + dh;
      union { unsigned u; __bf16 hh[2]; } cv;
      cv.u = pk2(qp[0], qp[1]);
      qf[c][2 * r] = cv.hh[0];
      qf[c][2 * r + 1] = cv.hh[1];
    }
  }

  float mrun[8], lrun[8];
  f32x8 oacc[8];
  #pragma unroll
  for (int r = 0; r < 8; ++r) { mrun[r] = -1e30f; lrun[r] = 0.f; }
  #pragma unroll
  for (int nf = 0; nf < 8; ++nf)
    #pragma unroll
    for (int r = 0; r < 8; ++r) oacc[nf][r] = 0.f;

  const float* kRow = K + h * DHEAD + lane * 4;
  const float* vRow = V + h * DHEAD + lane * 4;

  for (int kb = 0; kb < SEQ / 32; ++kb) {
    for (int r2 = 0; r2 < 32; ++r2) {
      float4 kv = *(const float4*)(kRow);
      float4 vv = *(const float4*)(vRow);
      kRow += DIMX;
      vRow += DIMX;
      *(uint2*)&Ks[r2 * KST + lane * 4] = pack4(kv);
      int d0 = lane * 4;
      Vt[(d0 + 0) * VTST + r2] = f2bf(vv.x);
      Vt[(d0 + 1) * VTST + r2] = f2bf(vv.y);
      Vt[(d0 + 2) * VTST + r2] = f2bf(vv.z);
      Vt[(d0 + 3) * VTST + r2] = f2bf(vv.w);
    }
    __syncthreads();

    f32x8 s0, s1;
    #pragma unroll
    for (int r = 0; r < 8; ++r) { s0[r] = 0.f; s1[r] = 0.f; }
    #pragma unroll
    for (int c = 0; c < 4; ++c) {
      bf16x16 kb0, kb1;
      #pragma unroll
      for (int r = 0; r < 8; ++r) {
        int kk = c * 32 + 2 * r + half * 16;
        ld_pair(&Ks[nl * KST + kk], kb0, 2 * r);
        ld_pair(&Ks[(16 + nl) * KST + kk], kb1, 2 * r);
      }
      s0 = __builtin_amdgcn_wmma_f32_16x16x32_bf16(false, qf[c], false, kb0, (short)0, s0, false, false);
      s1 = __builtin_amdgcn_wmma_f32_16x16x32_bf16(false, qf[c], false, kb1, (short)0, s1, false, false);
    }

    #pragma unroll
    for (int r = 0; r < 8; ++r) {
      float v0 = s0[r] * scale, v1 = s1[r] * scale;
      float mx = fmaxf(v0, v1);
      mx = fmaxf(mx, __shfl_xor(mx, 1, 32));
      mx = fmaxf(mx, __shfl_xor(mx, 2, 32));
      mx = fmaxf(mx, __shfl_xor(mx, 4, 32));
      mx = fmaxf(mx, __shfl_xor(mx, 8, 32));
      float mnew = fmaxf(mrun[r], mx);
      float alpha = __expf(mrun[r] - mnew);
      float p0 = __expf(v0 - mnew), p1 = __expf(v1 - mnew);
      float ps = p0 + p1;
      ps += __shfl_xor(ps, 1, 32);
      ps += __shfl_xor(ps, 2, 32);
      ps += __shfl_xor(ps, 4, 32);
      ps += __shfl_xor(ps, 8, 32);
      lrun[r] = lrun[r] * alpha + ps;
      mrun[r] = mnew;
      #pragma unroll
      for (int nf = 0; nf < 8; ++nf) oacc[nf][r] *= alpha;
      int prow = r + half * 8;
      Ps[prow * PST + nl] = f2bf(p0);
      Ps[prow * PST + 16 + nl] = f2bf(p1);
    }
    __syncthreads();

    bf16x16 pa;
    #pragma unroll
    for (int r = 0; r < 8; ++r)
      ld_pair(&Ps[nl * PST + a_kbase(r, half)], pa, 2 * r);
    #pragma unroll
    for (int nf = 0; nf < 8; ++nf) {
      bf16x16 vb;
      #pragma unroll
      for (int r = 0; r < 8; ++r)
        ld_pair(&Vt[(nf * 16 + nl) * VTST + 2 * r + half * 16], vb, 2 * r);
      oacc[nf] = __builtin_amdgcn_wmma_f32_16x16x32_bf16(false, pa, false, vb, (short)0, oacc[nf], false, false);
    }
    __syncthreads();
  }

  #pragma unroll
  for (int r = 0; r < 8; ++r) {
    float inv = 1.f / lrun[r];
    int row = q0 + r + half * 8;
    #pragma unroll
    for (int nf = 0; nf < 8; ++nf)
      O[(size_t)row * DIMX + h * DHEAD + nf * 16 + nl] = oacc[nf][r] * inv;
  }
}

// ---------------------------------------------------------------------------
// Elementwise / reduction helpers
// ---------------------------------------------------------------------------
__global__ void silu_k(const float* __restrict__ x, float* __restrict__ y, int n) {
  int i = blockIdx.x * blockDim.x + threadIdx.x;
  if (i < n) { float v = x[i]; y[i] = v / (1.f + __expf(-v)); }
}

__global__ void gemv_mod_k(const float* __restrict__ sa, const float* __restrict__ W,
                           const float* __restrict__ b, float* __restrict__ mod) {
  int j = blockIdx.x * blockDim.x + threadIdx.x;
  float acc = b[j];
  const float* w = W + j;
  for (int d = 0; d < DIMX; ++d) acc += sa[d] * w[(size_t)d * MOD6];
  mod[j] = acc;
}

__global__ __launch_bounds__(256)
void ln_mod_k(const float* __restrict__ X, const float* __restrict__ mod,
              int shOff, int scOff, float* __restrict__ Y) {
  __shared__ float rs[256], rq[256];
  int row = blockIdx.x, tid = threadIdx.x;
  float s = 0.f, q = 0.f;
  for (int j = tid; j < DIMX; j += 256) {
    float v = X[(size_t)row * DIMX + j];
    s += v; q += v * v;
  }
  rs[tid] = s; rq[tid] = q;
  __syncthreads();
  for (int off = 128; off > 0; off >>= 1) {
    if (tid < off) { rs[tid] += rs[tid + off]; rq[tid] += rq[tid + off]; }
    __syncthreads();
  }
  float mean = rs[0] / DIMX;
  float var  = rq[0] / DIMX - mean * mean;
  float rstd = rsqrtf(var + 1e-6f);
  for (int j = tid; j < DIMX; j += 256) {
    float v = X[(size_t)row * DIMX + j];
    Y[(size_t)row * DIMX + j] = (v - mean) * rstd * (1.f + mod[scOff + j]) + mod[shOff + j];
  }
}

__global__ __launch_bounds__(256)
void rms_k(float* __restrict__ X, const float* __restrict__ w) {
  __shared__ float rq[256];
  int row = blockIdx.x, tid = threadIdx.x;
  float q = 0.f;
  for (int j = tid; j < DIMX; j += 256) { float v = X[(size_t)row * DIMX + j]; q += v * v; }
  rq[tid] = q;
  __syncthreads();
  for (int off = 128; off > 0; off >>= 1) {
    if (tid < off) rq[tid] += rq[tid + off];
    __syncthreads();
  }
  float r = rsqrtf(rq[0] / DIMX + 1e-5f);
  for (int j = tid; j < DIMX; j += 256)
    X[(size_t)row * DIMX + j] = X[(size_t)row * DIMX + j] * r * w[j];
}

__global__ void residual_k(const float* __restrict__ ow, const float* __restrict__ tok,
                           const float* __restrict__ mod, float* __restrict__ img, int n) {
  int i = blockIdx.x * blockDim.x + threadIdx.x;
  if (i < n) { int j = i & (DIMX - 1); img[i] = mod[2 * DIMX + j] * ow[i] + tok[i]; }
}

__global__ void gate_k(const float* __restrict__ X, const float* __restrict__ gw,
                       float* __restrict__ comb) {
  int n = blockIdx.x * blockDim.x + threadIdx.x;
  if (n >= SEQ) return;
  float sc[NEXP] = {0.f, 0.f, 0.f, 0.f};
  for (int d = 0; d < DIMX; ++d) {
    float xv = X[(size_t)n * DIMX + d];
    #pragma unroll
    for (int e = 0; e < NEXP; ++e) sc[e] += xv * gw[e * DIMX + d];
  }
  float m = sc[0];
  #pragma unroll
  for (int e = 1; e < NEXP; ++e) m = fmaxf(m, sc[e]);
  float sum = 0.f, p[NEXP];
  #pragma unroll
  for (int e = 0; e < NEXP; ++e) { p[e] = __expf(sc[e] - m); sum += p[e]; }
  #pragma unroll
  for (int e = 0; e < NEXP; ++e) p[e] /= sum;
  int i0 = 0;
  #pragma unroll
  for (int e = 1; e < NEXP; ++e) if (p[e] > p[i0]) i0 = e;
  int i1 = -1;
  #pragma unroll
  for (int e = 0; e < NEXP; ++e) {
    if (e == i0) continue;
    if (i1 < 0 || p[e] > p[i1]) i1 = e;
  }
  #pragma unroll
  for (int e = 0; e < NEXP; ++e)
    comb[n * NEXP + e] = (e == i0 || e == i1) ? p[e] : 0.f;
}

__global__ void swiglu_k(float* __restrict__ H1, const float* __restrict__ H3, size_t n) {
  size_t i = (size_t)blockIdx.x * blockDim.x + threadIdx.x;
  if (i < n) { float v = H1[i]; H1[i] = (v / (1.f + __expf(-v))) * H3[i]; }
}

__global__ void accum_k(float* __restrict__ dst, const float* __restrict__ R,
                        const float* __restrict__ comb, int e, int zero, int n) {
  int i = blockIdx.x * blockDim.x + threadIdx.x;
  if (i < n) {
    int tok = i >> 11;   // / DIMX
    float w = comb[tok * NEXP + e];
    float prev = zero ? 0.f : dst[i];
    dst[i] = prev + w * R[i];
  }
}

__global__ void final_k(const float* __restrict__ routed, const float* __restrict__ shr,
                        const float* __restrict__ img, const float* __restrict__ mod,
                        float* __restrict__ out, int n) {
  int i = blockIdx.x * blockDim.x + threadIdx.x;
  if (i < n) {
    int j = i & (DIMX - 1);
    out[i] = mod[5 * DIMX + j] * (routed[i] + shr[i]) + img[i];
  }
}

// ---------------------------------------------------------------------------
extern "C" void kernel_launch(void* const* d_in, const int* in_sizes, int n_in,
                              void* d_out, int out_size, void* d_ws, size_t ws_size,
                              hipStream_t stream) {
  (void)in_sizes; (void)n_in; (void)out_size; (void)ws_size;
  const float* tokens = (const float*)d_in[0];
  const float* adaln  = (const float*)d_in[1];
  const float* W_ada  = (const float*)d_in[2];
  const float* b_ada  = (const float*)d_in[3];
  const float* Wq = (const float*)d_in[4];  const float* bq = (const float*)d_in[5];
  const float* Wk = (const float*)d_in[6];  const float* bk = (const float*)d_in[7];
  const float* Wv = (const float*)d_in[8];  const float* bv = (const float*)d_in[9];
  const float* qrw = (const float*)d_in[10];
  const float* krw = (const float*)d_in[11];
  const float* Wo = (const float*)d_in[12]; const float* bo = (const float*)d_in[13];
  const float* gw  = (const float*)d_in[14];
  const float* We1 = (const float*)d_in[15];
  const float* We3 = (const float*)d_in[16];
  const float* We2 = (const float*)d_in[17];
  const float* Ws1 = (const float*)d_in[18];
  const float* Ws3 = (const float*)d_in[19];
  const float* Ws2 = (const float*)d_in[20];
  float* out = (float*)d_out;

  float* ws = (float*)d_ws;
  size_t off = 0;
  float* MODp = ws + off; off += MOD6;
  float* SA   = ws + off; off += DIMX;
  float* COMB = ws + off; off += (size_t)SEQ * NEXP;
  float* XLN  = ws + off; off += (size_t)SEQ * DIMX;   // ln1 out, later ln2 out (y)
  float* QB   = ws + off; off += (size_t)SEQ * DIMX;   // q, later attn-proj out, later shared out
  float* KB   = ws + off; off += (size_t)SEQ * DIMX;   // k, later img
  float* VB   = ws + off; off += (size_t)SEQ * DIMX;   // v, later routed accumulator
  float* OB   = ws + off; off += (size_t)SEQ * DIMX;   // attn out, later per-expert out
  float* H1   = ws + off; off += (size_t)SEQ * HIDX;
  float* H3   = ws + off; off += (size_t)SEQ * HIDX;

  const int NT = SEQ * DIMX;
  dim3 gsq(DIMX / BLKN, SEQ / 64);

  // adaLN modulation
  silu_k<<<(DIMX + 255) / 256, 256, 0, stream>>>(adaln, SA, DIMX);
  gemv_mod_k<<<MOD6 / 256, 256, 0, stream>>>(SA, W_ada, b_ada, MODp);

  // attention branch
  ln_mod_k<<<SEQ, 256, 0, stream>>>(tokens, MODp, 0, DIMX, XLN);           // sh_msa, sc_msa
  gemm_bf16<<<gsq, 256, 0, stream>>>(XLN, DIMX, Wq, DIMX, bq, QB, DIMX, DIMX);
  gemm_bf16<<<gsq, 256, 0, stream>>>(XLN, DIMX, Wk, DIMX, bk, KB, DIMX, DIMX);
  gemm_bf16<<<gsq, 256, 0, stream>>>(XLN, DIMX, Wv, DIMX, bv, VB, DIMX, DIMX);
  rms_k<<<SEQ, 256, 0, stream>>>(QB, qrw);
  rms_k<<<SEQ, 256, 0, stream>>>(KB, krw);
  attn_fa<<<dim3(SEQ / 16, NHEAD), 32, 0, stream>>>(QB, KB, VB, OB);
  gemm_bf16<<<gsq, 256, 0, stream>>>(OB, DIMX, Wo, DIMX, bo, QB, DIMX, DIMX);
  residual_k<<<NT / 256, 256, 0, stream>>>(QB, tokens, MODp, KB, NT);      // img in KB

  // MoE branch
  ln_mod_k<<<SEQ, 256, 0, stream>>>(KB, MODp, 3 * DIMX, 4 * DIMX, XLN);    // sh_mlp, sc_mlp
  gate_k<<<SEQ / 256, 256, 0, stream>>>(XLN, gw, COMB);

  dim3 gh(HIDX / BLKN, SEQ / 64);
  for (int e = 0; e < NEXP; ++e) {
    gemm_bf16<<<gh, 256, 0, stream>>>(XLN, DIMX, We1 + (size_t)e * DIMX * HIDX, HIDX, nullptr, H1, HIDX, DIMX);
    gemm_bf16<<<gh, 256, 0, stream>>>(XLN, DIMX, We3 + (size_t)e * DIMX * HIDX, HIDX, nullptr, H3, HIDX, DIMX);
    swiglu_k<<<(unsigned)(((size_t)SEQ * HIDX + 255) / 256), 256, 0, stream>>>(H1, H3, (size_t)SEQ * HIDX);
    gemm_bf16<<<gsq, 256, 0, stream>>>(H1, HIDX, We2 + (size_t)e * HIDX * DIMX, DIMX, nullptr, OB, DIMX, HIDX);
    accum_k<<<NT / 256, 256, 0, stream>>>(VB, OB, COMB, e, e == 0 ? 1 : 0, NT);
  }

  // shared expert
  dim3 gs(HSX / BLKN, SEQ / 64);
  gemm_bf16<<<gs, 256, 0, stream>>>(XLN, DIMX, Ws1, HSX, nullptr, H1, HSX, DIMX);
  gemm_bf16<<<gs, 256, 0, stream>>>(XLN, DIMX, Ws3, HSX, nullptr, H3, HSX, DIMX);
  swiglu_k<<<(unsigned)(((size_t)SEQ * HSX + 255) / 256), 256, 0, stream>>>(H1, H3, (size_t)SEQ * HSX);
  gemm_bf16<<<gsq, 256, 0, stream>>>(H1, HSX, Ws2, DIMX, nullptr, QB, DIMX, HSX);

  final_k<<<NT / 256, 256, 0, stream>>>(VB, QB, KB, MODp, out, NT);
}
